// NeighborhoodEmbedding_5360119186123
// MI455X (gfx1250) — matrix operand order repaired and checked by reading, not compiled, not run on hardware
//
#include <hip/hip_runtime.h>

#define BATCH   4
#define NPTS    4096
#define KNN     16
#define COUT    256
#define NROWS   (BATCH*NPTS*KNN)     // 262144
#define ROWTILES (NROWS/16)          // 16384 (== number of points)
#define KTILES  (COUT/32)            // 8
#define NTILES  (COUT/16)            // 16
#define EPSBN   1e-5f

typedef __attribute__((ext_vector_type(16))) _Float16 v16h;
typedef __attribute__((ext_vector_type(8)))  _Float16 v8h;
typedef __attribute__((ext_vector_type(8)))  float    v8f;

// ---------------- workspace layout (bytes) ----------------
#define OFF_IDX   ((size_t)0)                                          // 1 MB int idx
#define OFF_COMB  (OFF_IDX  + (size_t)NROWS*sizeof(int))               // 6 MB f32 comb
#define OFF_H1PK  (OFF_COMB + (size_t)NROWS*6*sizeof(float))           // 128 MB f16 packed A
#define OFF_W2PK  (OFF_H1PK + (size_t)NROWS*COUT*2)                    // 128 KB f16 packed B
#define OFF_GMOM  (OFF_W2PK + (size_t)COUT*COUT*2)                     // 27 f32 moments
#define OFF_SC1   (OFF_GMOM + 1024)
#define OFF_SH1   (OFF_SC1  + 1024)
#define OFF_ST2   (OFF_SH1  + 1024)                                    // 512 f32 stats2
#define OFF_SC2   (OFF_ST2  + 2048)
#define OFF_SH2   (OFF_SC2  + 1024)

// ---------------- zero the global accumulators ----------------
__global__ void zero_stats_kernel(float* gmom, float* gstat2) {
  const int t = threadIdx.x;
  if (t < 27) gmom[t] = 0.f;
  gstat2[t] = 0.f;
  gstat2[COUT + t] = 0.f;
}

// ---------------- KNN: one thread per query point, LDS-tiled candidates ----------------
__global__ void knn_kernel(const float* __restrict__ xyz, int* __restrict__ idx) {
  __shared__ float sx[256], sy[256], sz[256];
  const int tid = threadIdx.x;
  const int b = blockIdx.x >> 4;                        // 16 blocks per batch
  const int gq = b * NPTS + ((blockIdx.x & 15) << 8) + tid;
  const float qx = xyz[3*gq+0], qy = xyz[3*gq+1], qz = xyz[3*gq+2];
  float bd[KNN]; int bi[KNN];
#pragma unroll
  for (int j = 0; j < KNN; ++j) { bd[j] = 3.4e38f; bi[j] = 0; }
  for (int tile = 0; tile < NPTS; tile += 256) {
    __syncthreads();
    const int cp = b * NPTS + tile + tid;
    sx[tid] = xyz[3*cp+0]; sy[tid] = xyz[3*cp+1]; sz[tid] = xyz[3*cp+2];
    __syncthreads();
    for (int jj = 0; jj < 256; ++jj) {
      const float dx = qx - sx[jj], dy = qy - sy[jj], dz = qz - sz[jj];
      const float d = fmaf(dx, dx, fmaf(dy, dy, dz*dz));
      if (d < bd[KNN-1]) {                 // rare path: branchless sorted insert
        const int cand = b * NPTS + tile + jj;
#pragma unroll
        for (int j = KNN-1; j > 0; --j) {
          const bool sh = d < bd[j-1];
          const bool he = (!sh) && (d < bd[j]);
          bd[j] = sh ? bd[j-1] : (he ? d    : bd[j]);
          bi[j] = sh ? bi[j-1] : (he ? cand : bi[j]);
        }
        if (d < bd[0]) { bd[0] = d; bi[0] = cand; }
      }
    }
  }
#pragma unroll
  for (int j = 0; j < KNN; ++j) idx[gq*KNN + j] = bi[j];
}

// ---------------- build comb rows + accumulate 1st/2nd moments (for analytic BN1) ----------
__global__ void comb_kernel(const float* __restrict__ xyz, const float* __restrict__ feat,
                            const int* __restrict__ idx, float* __restrict__ comb,
                            float* __restrict__ gmom) {
  __shared__ float smom[27];
  const int tid = threadIdx.x;
  if (tid < 27) smom[tid] = 0.f;
  __syncthreads();
  const int r = blockIdx.x * 256 + tid;
  const int p = r >> 4;
  const int g = idx[r];
  float v[6];
  v[0] = xyz[3*g+0] - xyz[3*p+0];
  v[1] = xyz[3*g+1] - xyz[3*p+1];
  v[2] = xyz[3*g+2] - xyz[3*p+2];
  v[3] = feat[3*g+0] - feat[3*p+0];
  v[4] = feat[3*g+1] - feat[3*p+1];
  v[5] = feat[3*g+2] - feat[3*p+2];
#pragma unroll
  for (int j = 0; j < 6; ++j) comb[r*6 + j] = v[j];
  float mm[27];
#pragma unroll
  for (int j = 0; j < 6; ++j) mm[j] = v[j];
  int t = 6;
#pragma unroll
  for (int j = 0; j < 6; ++j)
#pragma unroll
    for (int k = j; k < 6; ++k) { mm[t] = v[j]*v[k]; ++t; }
#pragma unroll
  for (int off = 16; off > 0; off >>= 1)
#pragma unroll
    for (int q = 0; q < 27; ++q) mm[q] += __shfl_xor(mm[q], off);
  if ((tid & 31) == 0)
#pragma unroll
    for (int q = 0; q < 27; ++q) atomicAdd(&smom[q], mm[q]);
  __syncthreads();
  if (tid < 27) atomicAdd(&gmom[tid], smom[tid]);
}

// ---------------- BN1 scale/shift from analytic moments ----------------
__global__ void bn1_params_kernel(const float* __restrict__ gmom, const float* __restrict__ W1,
                                  const float* __restrict__ b1, const float* __restrict__ g1,
                                  const float* __restrict__ beta1,
                                  float* __restrict__ sc1, float* __restrict__ sh1) {
  const int c = threadIdx.x;
  const float inv = 1.f / (float)NROWS;
  float w[6], mu[6];
#pragma unroll
  for (int j = 0; j < 6; ++j) { w[j] = W1[j*COUT + c]; mu[j] = gmom[j] * inv; }
  float ms = 0.f;
#pragma unroll
  for (int j = 0; j < 6; ++j) ms = fmaf(mu[j], w[j], ms);
  float es2 = 0.f;
  int t = 6;
#pragma unroll
  for (int j = 0; j < 6; ++j)
#pragma unroll
    for (int k = j; k < 6; ++k) {
      const float coef = (j == k) ? 1.f : 2.f;
      es2 = fmaf(coef * gmom[t] * inv, w[j]*w[k], es2);
      ++t;
    }
  const float var  = es2 - ms*ms;
  const float mean = ms + b1[c];
  const float s = g1[c] * rsqrtf(var + EPSBN);
  sc1[c] = s;
  sh1[c] = beta1[c] - mean * s;
}

// ---------------- GEMM1 (K=6, VALU) + BN1 + ReLU, write f16 in WMMA A-layout ------------
__global__ void pack_h1_kernel(const float* __restrict__ comb, const float* __restrict__ W1,
                               const float* __restrict__ b1, const float* __restrict__ sc1,
                               const float* __restrict__ sh1, _Float16* __restrict__ h1pk) {
  const int T = blockIdx.x * 256 + threadIdx.x;
  const int r  = T >> 5;           // row (32 consecutive threads share one row)
  const int u  = T & 31;
  const int kt = u >> 2;           // K-tile (32 channels)
  const int g  = (u >> 1) & 1;     // lane group (k>>4)&1
  const int h  = u & 1;            // element half (k>>3)&1
  float cb[6];
#pragma unroll
  for (int j = 0; j < 6; ++j) cb[j] = comb[r*6 + j];
  const int c0 = kt*32 + g*16 + h*8;
  v8h o;
#pragma unroll
  for (int i = 0; i < 8; ++i) {
    const int c = c0 + i;
    float y = b1[c];
#pragma unroll
    for (int j = 0; j < 6; ++j) y = fmaf(cb[j], W1[j*COUT + c], y);
    const float val = fmaxf(fmaf(y, sc1[c], sh1[c]), 0.f);
    o[i] = (_Float16)val;
  }
  const int rt = r >> 4, m = r & 15;
  const int lane = m + (g << 4);
  *(v8h*)(h1pk + ((size_t)(rt*KTILES + kt)*32 + lane)*16 + h*8) = o;
}

// ---------------- pack W2 into WMMA B-layout (lane holds 16 consecutive K of one col) ----
__global__ void pack_w2_kernel(const float* __restrict__ W2, _Float16* __restrict__ w2pk) {
  const int T = blockIdx.x * 256 + threadIdx.x;   // 4096 threads
  const int lane = T & 31;
  const int tile = T >> 5;
  const int kt = tile & 7;
  const int nt = tile >> 3;
  const int n = nt*16 + (lane & 15);
  const int kbase = kt*32 + (lane >> 4)*16;
  _Float16* dst = w2pk + ((size_t)(nt*KTILES + kt)*32 + lane)*16;
#pragma unroll
  for (int e = 0; e < 16; ++e) dst[e] = (_Float16)W2[(kbase + e)*COUT + n];
}

// ---------------- GEMM2 pass 1: WMMA, accumulate per-column sum / sumsq -----------------
__global__ void gemm2_stats_kernel(const _Float16* __restrict__ h1pk,
                                   const _Float16* __restrict__ w2pk,
                                   const float* __restrict__ b2,
                                   float* __restrict__ gstat2) {
  __shared__ float ssum[COUT], ssq[COUT];
  const int tid = threadIdx.x;
  ssum[tid] = 0.f; ssq[tid] = 0.f;
  __syncthreads();
  const int rt = blockIdx.x*8 + (tid >> 5);       // one 16-row tile per wave
  const int lane = tid & 31;
  const _Float16* aB = h1pk + (size_t)rt*KTILES*512 + lane*16;
  const int cl = lane & 15;
  for (int nt = 0; nt < NTILES; ++nt) {
    const _Float16* bB = w2pk + (size_t)nt*KTILES*512 + lane*16;
    v8f acc = {0.f,0.f,0.f,0.f,0.f,0.f,0.f,0.f};
#pragma unroll
    for (int kt = 0; kt < KTILES; ++kt) {
      v16h a  = *(const v16h*)(aB + kt*512);
      v16h bm = *(const v16h*)(bB + kt*512);
      acc = __builtin_amdgcn_wmma_f32_16x16x32_f16(false, a, false, bm,
                                                   (short)0, acc, false, false);
    }
    const int c = nt*16 + cl;
    const float bc = b2[c];
    float s = 0.f, q = 0.f;
#pragma unroll
    for (int j = 0; j < 8; ++j) { const float v = acc[j] + bc; s += v; q = fmaf(v, v, q); }
    atomicAdd(&ssum[c], s);
    atomicAdd(&ssq[c], q);
  }
  __syncthreads();
  atomicAdd(&gstat2[tid],        ssum[tid]);
  atomicAdd(&gstat2[COUT + tid], ssq[tid]);
}

// ---------------- BN2 scale/shift ----------------
__global__ void bn2_params_kernel(const float* __restrict__ gstat2, const float* __restrict__ g2,
                                  const float* __restrict__ beta2,
                                  float* __restrict__ sc2, float* __restrict__ sh2) {
  const int c = threadIdx.x;
  const float inv = 1.f / (float)NROWS;
  const float m   = gstat2[c] * inv;
  const float var = gstat2[COUT + c] * inv - m*m;
  const float s = g2[c] * rsqrtf(var + EPSBN);
  sc2[c] = s;
  sh2[c] = beta2[c] - m * s;
}

// ---------------- GEMM2 pass 2: WMMA + BN2 + ReLU + max over K (one point per 16-row tile)
__global__ void gemm2_max_kernel(const _Float16* __restrict__ h1pk,
                                 const _Float16* __restrict__ w2pk,
                                 const float* __restrict__ b2,
                                 const float* __restrict__ sc2,
                                 const float* __restrict__ sh2,
                                 float* __restrict__ out) {
  const int tid = threadIdx.x;
  const int rt = blockIdx.x*8 + (tid >> 5);
  const int lane = tid & 31;
  const _Float16* aB = h1pk + (size_t)rt*KTILES*512 + lane*16;
  const int cl = lane & 15;
  for (int nt = 0; nt < NTILES; ++nt) {
    const _Float16* bB = w2pk + (size_t)nt*KTILES*512 + lane*16;
    v8f acc = {0.f,0.f,0.f,0.f,0.f,0.f,0.f,0.f};
#pragma unroll
    for (int kt = 0; kt < KTILES; ++kt) {
      v16h a  = *(const v16h*)(aB + kt*512);
      v16h bm = *(const v16h*)(bB + kt*512);
      acc = __builtin_amdgcn_wmma_f32_16x16x32_f16(false, a, false, bm,
                                                   (short)0, acc, false, false);
    }
    const int c = nt*16 + cl;
    const float bc = b2[c], sc = sc2[c], sh = sh2[c];
    float mx = 0.f;                       // relu floor
#pragma unroll
    for (int j = 0; j < 8; ++j) {
      const float v = fmaf(acc[j] + bc, sc, sh);
      mx = fmaxf(mx, fmaxf(v, 0.f));
    }
    mx = fmaxf(mx, __shfl_xor(mx, 16));   // combine the two 8-row halves
    if (lane < 16) out[(size_t)rt*COUT + c] = mx;
  }
}

extern "C" void kernel_launch(void* const* d_in, const int* in_sizes, int n_in,
                              void* d_out, int out_size, void* d_ws, size_t ws_size,
                              hipStream_t stream) {
  (void)in_sizes; (void)n_in; (void)out_size; (void)ws_size;
  const float* xyz   = (const float*)d_in[0];
  const float* feat  = (const float*)d_in[1];
  const float* W1    = (const float*)d_in[2];
  const float* b1    = (const float*)d_in[3];
  const float* g1    = (const float*)d_in[4];
  const float* beta1 = (const float*)d_in[5];
  const float* W2    = (const float*)d_in[6];
  const float* b2    = (const float*)d_in[7];
  const float* g2    = (const float*)d_in[8];
  const float* beta2 = (const float*)d_in[9];
  float* out = (float*)d_out;

  char* ws = (char*)d_ws;
  int*      idx    = (int*)(ws + OFF_IDX);
  float*    comb   = (float*)(ws + OFF_COMB);
  _Float16* h1pk   = (_Float16*)(ws + OFF_H1PK);
  _Float16* w2pk   = (_Float16*)(ws + OFF_W2PK);
  float*    gmom   = (float*)(ws + OFF_GMOM);
  float*    sc1    = (float*)(ws + OFF_SC1);
  float*    sh1    = (float*)(ws + OFF_SH1);
  float*    gstat2 = (float*)(ws + OFF_ST2);
  float*    sc2    = (float*)(ws + OFF_SC2);
  float*    sh2    = (float*)(ws + OFF_SH2);

  zero_stats_kernel<<<1, 256, 0, stream>>>(gmom, gstat2);
  knn_kernel<<<64, 256, 0, stream>>>(xyz, idx);
  comb_kernel<<<NROWS/256, 256, 0, stream>>>(xyz, feat, idx, comb, gmom);
  bn1_params_kernel<<<1, 256, 0, stream>>>(gmom, W1, b1, g1, beta1, sc1, sh1);
  pack_w2_kernel<<<16, 256, 0, stream>>>(W2, w2pk);
  pack_h1_kernel<<<(NROWS/256)*32, 256, 0, stream>>>(comb, W1, b1, sc1, sh1, h1pk);
  gemm2_stats_kernel<<<ROWTILES/8, 256, 0, stream>>>(h1pk, w2pk, b2, gstat2);
  bn2_params_kernel<<<1, 256, 0, stream>>>(gstat2, g2, beta2, sc2, sh2);
  gemm2_max_kernel<<<ROWTILES/8, 256, 0, stream>>>(h1pk, w2pk, b2, sc2, sh2, out);
}
